// SpatiotemporalTransformerGAT_25675314495658
// MI455X (gfx1250) — matrix operand order repaired and checked by reading, not compiled, not run on hardware
//
#include <hip/hip_runtime.h>
#include <hip/hip_bf16.h>
#include <math.h>

// ---------------------------------------------------------------------------
// SpatiotemporalTransformerGAT for MI455X (gfx1250, wave32, WMMA).
// All dense matmuls run through v_wmma_f32_16x16x32_f16 (f16 in, f32 acc).
// ---------------------------------------------------------------------------

typedef __attribute__((ext_vector_type(16))) _Float16 v16h;
typedef __attribute__((ext_vector_type(8)))  float    v8f;

#define B_    4
#define T_    20
#define M_    10000
#define HOR_  4
#define DM    32
#define NH    4
#define DH    8
#define DFF_  128
#define HID_  64
#define GH    4
#define GOUT_ 16
#define E_    320000
#define E2_   (E_ + M_)
#define S_    (B_ * M_)          // 40000 sequences / nodes

// ---------------- workspace layout (bytes) ----------------
#define OFF_X      ((size_t)0)           // 40000*20*32 f32 = 102,400,000
#define OFF_XSKIP  ((size_t)102400000)   // 40000*64 f32    =  10,240,000
#define OFF_NODE   ((size_t)112640000)   // 40000*32 f32    =   5,120,000
#define OFF_XW     ((size_t)117760000)   // 40000*64 f32    =  10,240,000
#define OFF_ASRC   ((size_t)128000000)   // 40000*4 f32     =     640,000
#define OFF_ADST   ((size_t)128640000)   // 40000*4 f32     =     640,000
#define OFF_AMAX   ((size_t)129280000)   // 40000*4 i32     =     640,000
#define OFF_DEN    ((size_t)129920000)   // 40000*4 f32     =     640,000
#define OFF_GOUT   ((size_t)130560000)   // 40000*64 f32    =  10,240,000
#define OFF_XIN    ((size_t)140800000)   // 40000*64 f32    =  10,240,000
#define OFF_XS     ((size_t)151040000)   // 40000*64 f32    =  10,240,000

// ---------------- WMMA fragment helpers (contiguous k order, k = 16*half + s) ----------------
// A 16x32 f16: lanes0-15 M=lane, Vv: K=2v,2v+1 ; lanes16-31 K += 16
//   -> per lane one contiguous 16-half run (K=0..15 [+16 hi]) = 2x ds_load_b128
// B 32x16 f16: lanes0-15 N=lane, Vv: K=2v,2v+1 ; lanes16-31 K += 16
//   -> per lane one contiguous 16-half run = 2x ds_load_b128
// C/D 16x16 f32: VGPR r -> M = r + 8*(lane>=16), N = lane&15
__device__ __forceinline__ v8f wmma_gemm_tile(const _Float16* A, int lda, int m0,
                                              const _Float16* W, int ldw, int n0,
                                              int k_total) {
  const int lane = threadIdx.x & 31;
  const int half = lane >> 4;
  const int mrow = lane & 15;   // A row within tile
  const int ncol = lane & 15;   // B col within tile
  v8f acc = {};
  for (int k0 = 0; k0 < k_total; k0 += 32) {
    v16h a, b;
#pragma unroll
    for (int v = 0; v < 8; ++v) {
      int ka = 2 * v + 16 * half;
      int kb = 2 * v + 16 * half;
      a[2 * v]     = A[(m0 + mrow) * lda + k0 + ka];
      a[2 * v + 1] = A[(m0 + mrow) * lda + k0 + ka + 1];
      b[2 * v]     = W[(n0 + ncol) * ldw + k0 + kb];     // W row-major (N_out, K)
      b[2 * v + 1] = W[(n0 + ncol) * ldw + k0 + kb + 1];
    }
    acc = __builtin_amdgcn_wmma_f32_16x16x32_f16(false, a, false, b,
                                                 (short)0, acc, false, false);
  }
  return acc;
}

// bias value hoisted: exactly one ds_load of bias per tile store (no per-element branches)
__device__ __forceinline__ void store_tile_f32_bias(v8f acc, float* D, int ldd,
                                                    int m0, int n0, const float* bias) {
  const int lane = threadIdx.x & 31;
  const int half = lane >> 4;
  const int n = lane & 15;
  const float bv = bias[n0 + n];
#pragma unroll
  for (int r = 0; r < 8; ++r)
    D[(m0 + r + 8 * half) * ldd + n0 + n] = acc[r] + bv;
}

__device__ __forceinline__ void store_tile_f32_nobias(v8f acc, float* D, int ldd,
                                                      int m0, int n0) {
  const int lane = threadIdx.x & 31;
  const int half = lane >> 4;
  const int n = lane & 15;
#pragma unroll
  for (int r = 0; r < 8; ++r)
    D[(m0 + r + 8 * half) * ldd + n0 + n] = acc[r];
}

__device__ __forceinline__ float gelu_exact(float x) {
  return 0.5f * x * (1.0f + erff(x * 0.70710678118f));
}

__device__ __forceinline__ void store_tile_gelu_f16(v8f acc, _Float16* D, int ldd,
                                                    int m0, int n0, const float* bias) {
  const int lane = threadIdx.x & 31;
  const int half = lane >> 4;
  const int n = lane & 15;
  const float bv = bias[n0 + n];
#pragma unroll
  for (int r = 0; r < 8; ++r)
    D[(m0 + r + 8 * half) * ldd + n0 + n] = (_Float16)gelu_exact(acc[r] + bv);
}

// order-preserving int encoding of float for atomicMax
__device__ __forceinline__ int   enc_f(float f) { int i = __float_as_int(f); return i < 0 ? (i ^ 0x7fffffff) : i; }
__device__ __forceinline__ float dec_f(int i)   { return __int_as_float(i < 0 ? (i ^ 0x7fffffff) : i); }

// ---------------------------------------------------------------------------
// 1) input embedding + positional encoding : x[s][t][d]
// ---------------------------------------------------------------------------
__global__ void embed_kernel(const float* __restrict__ X, const float* __restrict__ iw,
                             const float* __restrict__ ib, float* __restrict__ xg) {
  int o = blockIdx.x * blockDim.x + threadIdx.x;
  if (o >= S_ * T_ * DM) return;
  int d = o % DM;
  int t = (o / DM) % T_;
  int s = o / (DM * T_);
  int b = s / M_, m = s % M_;
  float xv = X[(b * T_ + t) * M_ + m];
  int de = d & ~1;
  float arg = (float)t * __expf(-(float)de * 0.2878231366f);   // ln(1e4)/32
  float pe = (d & 1) ? __cosf(arg) : __sinf(arg);
  xg[o] = xv * iw[d] + ib[d] + pe;
}

// ---------------------------------------------------------------------------
// 2) skip branch: x_skip = LN(X[:, -1] * skip_w + skip_b)
// ---------------------------------------------------------------------------
__global__ void skip_kernel(const float* __restrict__ X, const float* __restrict__ sw,
                            const float* __restrict__ sb, const float* __restrict__ lg,
                            const float* __restrict__ lb, float* __restrict__ xskip) {
  int s = blockIdx.x * blockDim.x + threadIdx.x;
  if (s >= S_) return;
  int b = s / M_, m = s % M_;
  float xv = X[(b * T_ + (T_ - 1)) * M_ + m];
  float row[HID_];
  float mu = 0.f;
#pragma unroll
  for (int h = 0; h < HID_; ++h) { row[h] = xv * sw[h] + sb[h]; mu += row[h]; }
  mu *= (1.0f / HID_);
  float var = 0.f;
#pragma unroll
  for (int h = 0; h < HID_; ++h) { float t = row[h] - mu; var += t * t; }
  var *= (1.0f / HID_);
  float rs = rsqrtf(var + 1e-5f);
#pragma unroll
  for (int h = 0; h < HID_; ++h)
    xskip[s * HID_ + h] = (row[h] - mu) * rs * lg[h] + lb[h];
}

// ---------------------------------------------------------------------------
// 3) fused transformer encoder layer. 4 sequences (80 rows) per WG, all
//    activations resident in LDS, every GEMM via WMMA.  LDS = 54.8 KB.
// ---------------------------------------------------------------------------
struct EncP {
  const float *qkv_w, *qkv_b, *out_w, *out_b;
  const float *l1w, *l1b, *l2w, *l2b;
  const float *ln1g, *ln1b, *ln2g, *ln2b;
};

__global__ void __launch_bounds__(256) enc_layer_kernel(float* __restrict__ xg, EncP p) {
  __shared__ float                       xs[80][DM];       // 10,240 B residual/LN
  __shared__ _Float16                    xh[80][DM];       //  5,120 B WMMA A operand
  __shared__ __align__(16) unsigned char ubuf[30720];      // qkv f32[80][96] | h f16[80][128] + f f32[80][32]
  __shared__ _Float16                    wbuf[4096];       //  8,192 B staged weights (f16)
  __shared__ float                       bbuf[128];        //    512 B staged bias
  float*     qkvb = (float*)ubuf;
  _Float16*  hb   = (_Float16*)ubuf;
  float*     fb   = (float*)(ubuf + 20480);

  const int tid  = threadIdx.x;
  const int wave = tid >> 5;
  const int s0   = blockIdx.x * 4;             // 4 sequences per block

  // warm L2/L0 for later-stage weights (global_prefetch_b8); uniform address per wave
  __builtin_prefetch(p.out_w, 0, 1);
  __builtin_prefetch(p.l1w, 0, 1);
  __builtin_prefetch(p.l2w, 0, 1);

  // ---- stage 1: load x tile (contiguous) + Wqkv/bqkv ----
  for (int i = tid; i < 80 * DM; i += 256) {
    float v = xg[s0 * T_ * DM + i];
    xs[i / DM][i % DM] = v;
    xh[i / DM][i % DM] = (_Float16)v;
  }
  for (int i = tid; i < 3 * DM * DM; i += 256) wbuf[i] = (_Float16)p.qkv_w[i];
  for (int i = tid; i < 3 * DM;      i += 256) bbuf[i] = p.qkv_b[i];
  __syncthreads();

  // ---- stage 2: qkv = x @ Wqkv^T + b  (80x32 @ 32x96), 30 WMMA tiles ----
  for (int j = wave; j < 5 * 6; j += 8) {
    int mt = j / 6, nt = j % 6;
    v8f acc = wmma_gemm_tile(&xh[0][0], DM, mt * 16, wbuf, DM, nt * 16, DM);
    store_tile_f32_bias(acc, qkvb, 96, mt * 16, nt * 16, bbuf);
  }
  __syncthreads();

  // ---- stage 3: attention (per seq,head,query) -> o (f16) into xh; stage Wout ----
  for (int i = tid; i < DM * DM; i += 256) wbuf[i] = (_Float16)p.out_w[i];
  for (int i = tid; i < DM;      i += 256) bbuf[i] = p.out_b[i];
  for (int task = tid; task < 4 * NH * T_; task += 256) {
    int si = task / (NH * T_);
    int h  = (task / T_) % NH;
    int tq = task % T_;
    int r  = si * T_ + tq;
    const float* q = &qkvb[r * 96 + h * DH];
    float sc[T_];
    float mx = -1e30f;
#pragma unroll
    for (int tk = 0; tk < T_; ++tk) {
      const float* kk = &qkvb[(si * T_ + tk) * 96 + DM + h * DH];
      float s = 0.f;
#pragma unroll
      for (int c = 0; c < DH; ++c) s += q[c] * kk[c];
      s *= 0.35355339059f;                   // 1/sqrt(8)
      sc[tk] = s;
      mx = fmaxf(mx, s);
    }
    float den = 0.f;
#pragma unroll
    for (int tk = 0; tk < T_; ++tk) { sc[tk] = __expf(sc[tk] - mx); den += sc[tk]; }
    float inv = 1.0f / den;
#pragma unroll
    for (int c = 0; c < DH; ++c) {
      float o = 0.f;
#pragma unroll
      for (int tk = 0; tk < T_; ++tk)
        o += sc[tk] * qkvb[(si * T_ + tk) * 96 + 2 * DM + h * DH + c];
      xh[r][h * DH + c] = (_Float16)(o * inv);
    }
  }
  __syncthreads();

  // ---- stage 4: out-proj (80x32 @ 32x32) -> qkvb[.][0..31] ----
  for (int j = wave; j < 5 * 2; j += 8) {
    int mt = j / 2, nt = j % 2;
    v8f acc = wmma_gemm_tile(&xh[0][0], DM, mt * 16, wbuf, DM, nt * 16, DM);
    store_tile_f32_bias(acc, qkvb, 96, mt * 16, nt * 16, bbuf);
  }
  __syncthreads();

  // ---- stage 5: residual + LN1 -> xs, xh; stage W1/b1 ----
  for (int i = tid; i < DFF_ * DM; i += 256) wbuf[i] = (_Float16)p.l1w[i];
  for (int i = tid; i < DFF_;      i += 256) bbuf[i] = p.l1b[i];
  if (tid < 80) {
    float row[DM];
    float mu = 0.f;
#pragma unroll
    for (int d = 0; d < DM; ++d) { row[d] = xs[tid][d] + qkvb[tid * 96 + d]; mu += row[d]; }
    mu *= (1.0f / DM);
    float var = 0.f;
#pragma unroll
    for (int d = 0; d < DM; ++d) { float t = row[d] - mu; var += t * t; }
    var *= (1.0f / DM);
    float rs = rsqrtf(var + 1e-5f);
#pragma unroll
    for (int d = 0; d < DM; ++d) {
      float y = (row[d] - mu) * rs * p.ln1g[d] + p.ln1b[d];
      xs[tid][d] = y;
      xh[tid][d] = (_Float16)y;
    }
  }
  __syncthreads();

  // ---- stage 6: FFN1 (80x32 @ 32x128) + gelu -> hb (f16) ----
  for (int j = wave; j < 5 * 8; j += 8) {
    int mt = j / 8, nt = j % 8;
    v8f acc = wmma_gemm_tile(&xh[0][0], DM, mt * 16, wbuf, DM, nt * 16, DM);
    store_tile_gelu_f16(acc, hb, DFF_, mt * 16, nt * 16, bbuf);
  }
  __syncthreads();

  // ---- stage 7: stage W2/b2 ----
  for (int i = tid; i < DM * DFF_; i += 256) wbuf[i] = (_Float16)p.l2w[i];
  for (int i = tid; i < DM;        i += 256) bbuf[i] = p.l2b[i];
  __syncthreads();

  // ---- stage 8: FFN2 (80x128 @ 128x32, K=128 -> 4 WMMA k-steps) -> fb ----
  for (int j = wave; j < 5 * 2; j += 8) {
    int mt = j / 2, nt = j % 2;
    v8f acc = wmma_gemm_tile(hb, DFF_, mt * 16, wbuf, DFF_, nt * 16, DFF_);
    store_tile_f32_bias(acc, fb, DM, mt * 16, nt * 16, bbuf);
  }
  __syncthreads();

  // ---- stage 9: residual + LN2 -> global x ----
  if (tid < 80) {
    float row[DM];
    float mu = 0.f;
#pragma unroll
    for (int d = 0; d < DM; ++d) { row[d] = xs[tid][d] + fb[tid * DM + d]; mu += row[d]; }
    mu *= (1.0f / DM);
    float var = 0.f;
#pragma unroll
    for (int d = 0; d < DM; ++d) { float t = row[d] - mu; var += t * t; }
    var *= (1.0f / DM);
    float rs = rsqrtf(var + 1e-5f);
#pragma unroll
    for (int d = 0; d < DM; ++d)
      xg[s0 * T_ * DM + tid * DM + d] = (row[d] - mu) * rs * p.ln2g[d] + p.ln2b[d];
  }
}

// ---------------------------------------------------------------------------
// 4) node = LN(mean_t x)
// ---------------------------------------------------------------------------
__global__ void meanln_kernel(const float* __restrict__ xg, const float* __restrict__ g,
                              const float* __restrict__ b, float* __restrict__ node) {
  int s = blockIdx.x * blockDim.x + threadIdx.x;
  if (s >= S_) return;
  float row[DM];
#pragma unroll
  for (int d = 0; d < DM; ++d) row[d] = 0.f;
  for (int t = 0; t < T_; ++t) {
#pragma unroll
    for (int d = 0; d < DM; ++d) row[d] += xg[(s * T_ + t) * DM + d];
  }
  float mu = 0.f;
#pragma unroll
  for (int d = 0; d < DM; ++d) { row[d] *= (1.0f / T_); mu += row[d]; }
  mu *= (1.0f / DM);
  float var = 0.f;
#pragma unroll
  for (int d = 0; d < DM; ++d) { float t = row[d] - mu; var += t * t; }
  var *= (1.0f / DM);
  float rs = rsqrtf(var + 1e-5f);
#pragma unroll
  for (int d = 0; d < DM; ++d)
    node[s * DM + d] = (row[d] - mu) * rs * g[d] + b[d];
}

// ---------------------------------------------------------------------------
// 5) xw = node @ gat_w^T  (WMMA 40000x32 @ 32x64)  + per-head asrc/adst
// ---------------------------------------------------------------------------
__global__ void __launch_bounds__(256) gatw_kernel(const float* __restrict__ node,
                                                   const float* __restrict__ gw,
                                                   const float* __restrict__ att_src,
                                                   const float* __restrict__ att_dst,
                                                   float* __restrict__ xw,
                                                   float* __restrict__ asrc,
                                                   float* __restrict__ adst) {
  __shared__ _Float16 nh_[64][DM];
  __shared__ _Float16 gwh[HID_ * DM];
  __shared__ float    xwl[64][HID_];
  const int tid = threadIdx.x, wave = tid >> 5;
  const int r0 = blockIdx.x * 64;
  for (int i = tid; i < 64 * DM; i += 256) nh_[i / DM][i % DM] = (_Float16)node[r0 * DM + i];
  for (int i = tid; i < HID_ * DM; i += 256) gwh[i] = (_Float16)gw[i];
  __syncthreads();
  for (int j = wave; j < 16; j += 8) {
    int mt = j / 4, nt = j % 4;
    v8f acc = wmma_gemm_tile(&nh_[0][0], DM, mt * 16, gwh, DM, nt * 16, DM);
    store_tile_f32_nobias(acc, &xwl[0][0], HID_, mt * 16, nt * 16);
  }
  __syncthreads();
  for (int i = tid; i < 64 * HID_; i += 256) xw[r0 * HID_ + i] = xwl[i / HID_][i % HID_];
  {
    int row = tid >> 2, h = tid & 3;
    float s1 = 0.f, s2 = 0.f;
#pragma unroll
    for (int c = 0; c < GOUT_; ++c) {
      float v = xwl[row][h * GOUT_ + c];
      s1 += v * att_src[h * GOUT_ + c];
      s2 += v * att_dst[h * GOUT_ + c];
    }
    asrc[(r0 + row) * GH + h] = s1;
    adst[(r0 + row) * GH + h] = s2;
  }
}

// ---------------------------------------------------------------------------
// 6) GAT segment-softmax scatter (3 passes)
// ---------------------------------------------------------------------------
__global__ void amax_init_kernel(int* __restrict__ amax) {
  int i = blockIdx.x * blockDim.x + threadIdx.x;
  if (i < S_ * GH) amax[i] = (int)0x80000000;   // enc(-inf) lower bound
}

__device__ __forceinline__ void edge_decode(int t, int& b, int& h, int& e) {
  e = t % E2_; int q = t / E2_; h = q % GH; b = q / GH;
}

__device__ __forceinline__ float edge_logit(const int* eidx, const float* asrc,
                                            const float* adst, int b, int h, int e,
                                            int& s_, int& d_) {
  s_ = (e < E_) ? eidx[e] : (e - E_);
  d_ = (e < E_) ? eidx[E_ + e] : (e - E_);
  float a = asrc[(b * M_ + s_) * GH + h] + adst[(b * M_ + d_) * GH + h];
  return (a < 0.f) ? 0.2f * a : a;     // leaky_relu(., 0.2)
}

__global__ void edge_max_kernel(const int* __restrict__ eidx, const float* __restrict__ asrc,
                                const float* __restrict__ adst, int* __restrict__ amax) {
  int t = blockIdx.x * blockDim.x + threadIdx.x;
  if (t >= B_ * GH * E2_) return;
  int b, h, e, s_, d_;
  edge_decode(t, b, h, e);
  float a = edge_logit(eidx, asrc, adst, b, h, e, s_, d_);
  atomicMax(&amax[(b * M_ + d_) * GH + h], enc_f(a));
}

__global__ void edge_sum_kernel(const int* __restrict__ eidx, const float* __restrict__ asrc,
                                const float* __restrict__ adst, const int* __restrict__ amax,
                                float* __restrict__ den) {
  int t = blockIdx.x * blockDim.x + threadIdx.x;
  if (t >= B_ * GH * E2_) return;
  int b, h, e, s_, d_;
  edge_decode(t, b, h, e);
  float a = edge_logit(eidx, asrc, adst, b, h, e, s_, d_);
  int idx = (b * M_ + d_) * GH + h;
  atomicAdd(&den[idx], __expf(a - dec_f(amax[idx])));
}

__global__ void edge_acc_kernel(const int* __restrict__ eidx, const float* __restrict__ asrc,
                                const float* __restrict__ adst, const int* __restrict__ amax,
                                const float* __restrict__ den, const float* __restrict__ xw,
                                float* __restrict__ gout) {
  int t = blockIdx.x * blockDim.x + threadIdx.x;
  if (t >= B_ * GH * E2_) return;
  int b, h, e, s_, d_;
  edge_decode(t, b, h, e);
  float a = edge_logit(eidx, asrc, adst, b, h, e, s_, d_);
  int idx = (b * M_ + d_) * GH + h;
  float alpha = __expf(a - dec_f(amax[idx])) / (den[idx] + 1e-16f);
  const float* xr = &xw[(b * M_ + s_) * HID_ + h * GOUT_];
  float*       gr = &gout[(b * M_ + d_) * HID_ + h * GOUT_];
#pragma unroll
  for (int c = 0; c < GOUT_; ++c) atomicAdd(&gr[c], alpha * xr[c]);
}

// ---------------------------------------------------------------------------
// 7) g = gelu(LN(gout)); x_in = 0.8*g + 0.2*x_skip
// ---------------------------------------------------------------------------
__global__ void gatpost_kernel(const float* __restrict__ gout, const float* __restrict__ lg,
                               const float* __restrict__ lb, const float* __restrict__ xskip,
                               float* __restrict__ xin) {
  int s = blockIdx.x * blockDim.x + threadIdx.x;
  if (s >= S_) return;
  float row[HID_];
  float mu = 0.f;
#pragma unroll
  for (int h = 0; h < HID_; ++h) { row[h] = gout[s * HID_ + h]; mu += row[h]; }
  mu *= (1.0f / HID_);
  float var = 0.f;
#pragma unroll
  for (int h = 0; h < HID_; ++h) { float t = row[h] - mu; var += t * t; }
  var *= (1.0f / HID_);
  float rs = rsqrtf(var + 1e-5f);
#pragma unroll
  for (int h = 0; h < HID_; ++h) {
    float y = (row[h] - mu) * rs * lg[h] + lb[h];
    xin[s * HID_ + h] = 0.8f * gelu_exact(y) + 0.2f * xskip[s * HID_ + h];
  }
}

// ---------------------------------------------------------------------------
// 8) residual projection block (two 64x64 GEMMs + LN + gelu, WMMA, K=64)
// ---------------------------------------------------------------------------
struct ProjP { const float *l1b, *l1w, *l2b, *l2w, *lnb, *lng; };

__global__ void __launch_bounds__(256) proj_kernel(const float* __restrict__ in,
                                                   float* __restrict__ out, ProjP p) {
  __shared__ float    xsl[64][HID_];   // 16 KB
  __shared__ _Float16 ah[64][HID_];    //  8 KB
  __shared__ float    hl[64][HID_];    // 16 KB
  __shared__ _Float16 w1[HID_ * HID_]; //  8 KB
  __shared__ _Float16 w2[HID_ * HID_]; //  8 KB
  __shared__ float    pb[256];         //  1 KB : b1 | b2 | ln_g | ln_b
  const int tid = threadIdx.x, wave = tid >> 5;
  const int r0 = blockIdx.x * 64;
  __builtin_prefetch(p.l1w, 0, 1);
  __builtin_prefetch(p.l2w, 0, 1);
  for (int i = tid; i < 64 * HID_; i += 256) {
    float v = in[r0 * HID_ + i];
    xsl[i / HID_][i % HID_] = v;
    ah[i / HID_][i % HID_] = (_Float16)v;
  }
  for (int i = tid; i < HID_ * HID_; i += 256) { w1[i] = (_Float16)p.l1w[i]; w2[i] = (_Float16)p.l2w[i]; }
  if (tid < 64) {
    pb[tid] = p.l1b[tid];  pb[64 + tid] = p.l2b[tid];
    pb[128 + tid] = p.lng[tid]; pb[192 + tid] = p.lnb[tid];
  }
  __syncthreads();
  // h = xs @ l1^T + b1
  for (int j = wave; j < 16; j += 8) {
    int mt = j / 4, nt = j % 4;
    v8f acc = wmma_gemm_tile(&ah[0][0], HID_, mt * 16, w1, HID_, nt * 16, HID_);
    store_tile_f32_bias(acc, &hl[0][0], HID_, mt * 16, nt * 16, pb);
  }
  __syncthreads();
  // h = gelu(LN(h)) -> ah (f16)
  if (tid < 64) {
    float mu = 0.f;
#pragma unroll
    for (int d = 0; d < HID_; ++d) mu += hl[tid][d];
    mu *= (1.0f / HID_);
    float var = 0.f;
#pragma unroll
    for (int d = 0; d < HID_; ++d) { float t = hl[tid][d] - mu; var += t * t; }
    var *= (1.0f / HID_);
    float rs = rsqrtf(var + 1e-5f);
#pragma unroll
    for (int d = 0; d < HID_; ++d) {
      float y = (hl[tid][d] - mu) * rs * pb[128 + d] + pb[192 + d];
      ah[tid][d] = (_Float16)gelu_exact(y);
    }
  }
  __syncthreads();
  // out = h @ l2^T + b2 + xs   (residual fused into the WMMA store)
  for (int j = wave; j < 16; j += 8) {
    int mt = j / 4, nt = j % 4;
    v8f acc = wmma_gemm_tile(&ah[0][0], HID_, mt * 16, w2, HID_, nt * 16, HID_);
    const int lane = tid & 31, half = lane >> 4, n = lane & 15;
    const float bv = pb[64 + nt * 16 + n];
#pragma unroll
    for (int r = 0; r < 8; ++r) {
      int m = mt * 16 + r + 8 * half, c = nt * 16 + n;
      out[(r0 + m) * HID_ + c] = acc[r] + bv + xsl[m][c];
    }
  }
}

// ---------------------------------------------------------------------------
// 9) final head + autoregressive state update
// ---------------------------------------------------------------------------
__global__ void fin_kernel(const float* __restrict__ xs, float* __restrict__ xin,
                           const float* __restrict__ finw, const float* __restrict__ finb,
                           float* __restrict__ preds, int step) {
  int s = blockIdx.x * blockDim.x + threadIdx.x;
  if (s >= S_) return;
  float acc = finb[0];
#pragma unroll
  for (int d = 0; d < HID_; ++d) acc += xs[s * HID_ + d] * finw[d];
  preds[s * HOR_ + step] = acc;
#pragma unroll
  for (int d = 0; d < HID_; ++d)
    xin[s * HID_ + d] = 0.6f * xin[s * HID_ + d] + 0.4f * acc;
}

// ---------------------------------------------------------------------------
// launch — input indices follow JAX pytree (sorted-key) flattening:
// 0:X 1:edge_index 2:edge_weight 3:att_dst 4:att_src 5:fin.b 6:fin.w 7:gat_w
// 8:inp.b 9:inp.w  layers l: base=10+12l -> lin1.b,lin1.w,lin2.b,lin2.w,
//   ln1_b,ln1_g,ln2_b,ln2_g,out.b,out.w,qkv.b,qkv.w
// 34:lng_b 35:lng_g 36:lnt_b 37:lnt_g  proj p: base=38+6p -> l1.b,l1.w,l2.b,
//   l2.w,ln_b,ln_g   50:skip.b 51:skip.w 52:skipln_b 53:skipln_g
// ---------------------------------------------------------------------------
extern "C" void kernel_launch(void* const* d_in, const int* in_sizes, int n_in,
                              void* d_out, int out_size, void* d_ws, size_t ws_size,
                              hipStream_t stream) {
  (void)in_sizes; (void)n_in; (void)out_size; (void)ws_size;
  auto F = [&](int i) { return (const float*)d_in[i]; };
  const float* X      = F(0);
  const int*   eidx   = (const int*)d_in[1];
  const float* attd   = F(3);
  const float* atts   = F(4);
  const float* fin_b  = F(5);
  const float* fin_w  = F(6);
  const float* gat_w  = F(7);
  const float* inp_b  = F(8);
  const float* inp_w  = F(9);
  const float* lng_b  = F(34);
  const float* lng_g  = F(35);
  const float* lnt_b  = F(36);
  const float* lnt_g  = F(37);
  const float* skip_b = F(50);
  const float* skip_w = F(51);
  const float* sln_b  = F(52);
  const float* sln_g  = F(53);

  char* ws = (char*)d_ws;
  float* xbuf  = (float*)(ws + OFF_X);
  float* xskip = (float*)(ws + OFF_XSKIP);
  float* node  = (float*)(ws + OFF_NODE);
  float* xw    = (float*)(ws + OFF_XW);
  float* asrc  = (float*)(ws + OFF_ASRC);
  float* adst  = (float*)(ws + OFF_ADST);
  int*   amax  = (int*)  (ws + OFF_AMAX);
  float* den   = (float*)(ws + OFF_DEN);
  float* gout  = (float*)(ws + OFF_GOUT);
  float* xin   = (float*)(ws + OFF_XIN);
  float* xsb   = (float*)(ws + OFF_XS);
  float* preds = (float*)d_out;

  // embedding + skip branch
  embed_kernel<<<(S_ * T_ * DM + 255) / 256, 256, 0, stream>>>(X, inp_w, inp_b, xbuf);
  skip_kernel<<<(S_ + 255) / 256, 256, 0, stream>>>(X, skip_w, skip_b, sln_g, sln_b, xskip);

  // transformer encoder layers (fused, WMMA)
  for (int l = 0; l < 2; ++l) {
    int bs = 10 + 12 * l;
    EncP p;
    p.qkv_w = F(bs + 11); p.qkv_b = F(bs + 10);
    p.out_w = F(bs + 9);  p.out_b = F(bs + 8);
    p.l1w   = F(bs + 1);  p.l1b   = F(bs + 0);
    p.l2w   = F(bs + 3);  p.l2b   = F(bs + 2);
    p.ln1g  = F(bs + 5);  p.ln1b  = F(bs + 4);
    p.ln2g  = F(bs + 7);  p.ln2b  = F(bs + 6);
    enc_layer_kernel<<<S_ / 4, 256, 0, stream>>>(xbuf, p);
  }

  // temporal pooling + GAT projection
  meanln_kernel<<<(S_ + 255) / 256, 256, 0, stream>>>(xbuf, lnt_g, lnt_b, node);
  gatw_kernel<<<S_ / 64, 256, 0, stream>>>(node, gat_w, atts, attd, xw, asrc, adst);

  // GAT segment softmax + aggregation
  amax_init_kernel<<<(S_ * GH + 255) / 256, 256, 0, stream>>>(amax);
  hipMemsetAsync(ws + OFF_DEN, 0, (size_t)(640000 + 10240000), stream);  // den + gout
  const int etotal = B_ * GH * E2_;
  const int eblks = (etotal + 255) / 256;
  edge_max_kernel<<<eblks, 256, 0, stream>>>(eidx, asrc, adst, amax);
  edge_sum_kernel<<<eblks, 256, 0, stream>>>(eidx, asrc, adst, amax, den);
  edge_acc_kernel<<<eblks, 256, 0, stream>>>(eidx, asrc, adst, amax, den, xw, gout);
  gatpost_kernel<<<(S_ + 255) / 256, 256, 0, stream>>>(gout, lng_g, lng_b, xskip, xin);

  // horizon rollout
  ProjP pp[2];
  for (int i = 0; i < 2; ++i) {
    int bp = 38 + 6 * i;
    pp[i].l1b = F(bp + 0); pp[i].l1w = F(bp + 1);
    pp[i].l2b = F(bp + 2); pp[i].l2w = F(bp + 3);
    pp[i].lnb = F(bp + 4); pp[i].lng = F(bp + 5);
  }
  for (int step = 0; step < HOR_; ++step) {
    proj_kernel<<<S_ / 64, 256, 0, stream>>>(xin, xsb, pp[0]);
    proj_kernel<<<S_ / 64, 256, 0, stream>>>(xsb, xsb, pp[1]);
    fin_kernel<<<(S_ + 255) / 256, 256, 0, stream>>>(xsb, xin, fin_w, fin_b, preds, step);
  }
}
